// ImprovedGNN_75058848465364
// MI455X (gfx1250) — compile-verified, matrix operand order checked by reading, not run on hardware
//
#include <hip/hip_runtime.h>
#include <hip/hip_bf16.h>

// ---------------- problem constants (match reference) ----------------
#define NN        50000
#define EE        800000
#define IN_F      128
#define HID_C     64
#define OUT_C     64
#define HEADS     4
#define EPS_      1e-5f
#define NEG_SLOPE 0.2f

// ---------------- WMMA vector types (CDNA5, wave32) ----------------
typedef __attribute__((ext_vector_type(16))) __bf16 v16bf;
typedef __attribute__((ext_vector_type(8)))  float  v8f;

// ---------------- float atomic max via integer ordering trick ----------------
__device__ __forceinline__ void atomicMaxF(float* addr, float v) {
    if (v >= 0.0f) atomicMax((int*)addr, __float_as_int(v));
    else           atomicMin((unsigned int*)addr, __float_as_uint(v));
}

// ---------------- bf16 WMMA GEMM: C[nrows x mcols] = A[nrows x K] @ B ----------------
// TRANSB=false: B is [K x mcols] row-major.  TRANSB=true: B is [mcols x K] row-major (A @ B^T).
// Weights staged once per block into LDS (transposed, bf16) -> ds_load_b32 per fragment dword.
// A fragments held in registers across all column tiles.  One wave = one 16-row stripe.
template <int K, bool TRANSB>
__global__ void gemm_bf16_wmma(const float* __restrict__ A,
                               const float* __restrict__ B,
                               float* __restrict__ C,
                               int nrows, int mcols) {
    extern __shared__ __bf16 Bt[];                  // [mcols][K] bf16 (transposed weights)
    const int tid = threadIdx.x;

    // ---- cooperative stage: Bt[n*K + k] = bf16(TRANSB ? B[n*K+k] : B[k*mcols+n]) ----
    const int total = mcols * K;
    for (int idx = tid; idx < total; idx += blockDim.x) {
        float v;
        if (TRANSB) v = B[idx];                     // already [mcols][K]
        else { const int n = idx / K, k = idx - n * K; v = B[(size_t)k * mcols + n]; }
        Bt[idx] = (__bf16)v;
    }
    __syncthreads();

    const int lane = tid & 31;
    const int wave = tid >> 5;
    const int rowTile = blockIdx.x * (blockDim.x >> 5) + wave;
    if (rowTile * 16 >= nrows) return;              // wave-uniform exit (EXEC all-1s for WMMA)
    const int m  = lane & 15;
    const int hi = lane >> 4;
    const int arow = rowTile * 16 + m;

    // ---- load all A fragments for this 16-row stripe (registers, reused by every col tile) ----
    // ISA 16-bit A 16x32 layout: lane's K values are two contiguous 8-float runs:
    //   [k0 + hi*8 .. +7]  and  [k0 + 16 + hi*8 .. +7]   -> four aligned float4 loads
    constexpr int KS = K / 32;
    v16bf afrag[KS];
#pragma unroll
    for (int ks = 0; ks < KS; ++ks) {
        const float* ap = A + (size_t)arow * K + ks * 32 + hi * 8;
        const float4 x0 = *(const float4*)(ap);
        const float4 x1 = *(const float4*)(ap + 4);
        const float4 x2 = *(const float4*)(ap + 16);
        const float4 x3 = *(const float4*)(ap + 20);
        v16bf a;
        a[0]  = (__bf16)x0.x; a[1]  = (__bf16)x0.y; a[2]  = (__bf16)x0.z; a[3]  = (__bf16)x0.w;
        a[4]  = (__bf16)x1.x; a[5]  = (__bf16)x1.y; a[6]  = (__bf16)x1.z; a[7]  = (__bf16)x1.w;
        a[8]  = (__bf16)x2.x; a[9]  = (__bf16)x2.y; a[10] = (__bf16)x2.z; a[11] = (__bf16)x2.w;
        a[12] = (__bf16)x3.x; a[13] = (__bf16)x3.y; a[14] = (__bf16)x3.z; a[15] = (__bf16)x3.w;
        afrag[ks] = a;
    }

    // ---- sweep all column tiles, B fragments from LDS ----
    const int nColTiles = mcols >> 4;
    for (int ct = 0; ct < nColTiles; ++ct) {
        const int bcol = ct * 16 + m;               // B fragment: lane holds column n
        v8f acc = {0.f, 0.f, 0.f, 0.f, 0.f, 0.f, 0.f, 0.f};
#pragma unroll
        for (int ks = 0; ks < KS; ++ks) {
            union { v16bf v; unsigned u[8]; } b;
            const unsigned* bp = (const unsigned*)&Bt[(size_t)bcol * K + ks * 32];
#pragma unroll
            for (int i = 0; i < 8; ++i) {
                const int kk = (i < 4) ? (hi * 8 + 2 * i) : (16 + hi * 8 + 2 * (i - 4));
                b.u[i] = bp[kk >> 1];               // one aligned 32-bit LDS load = bf16 pair
            }
            acc = __builtin_amdgcn_wmma_f32_16x16x32_bf16(false, afrag[ks], false, b.v,
                                                          (short)0, acc, false, false);
        }
        // C/D layout: VGPR r holds M = r + 8*hi, N = lane&15
#pragma unroll
        for (int r = 0; r < 8; ++r)
            C[(size_t)(rowTile * 16 + hi * 8 + r) * mcols + ct * 16 + m] = acc[r];
    }
}

// ---------------- attention coefficients: a_src/a_dst = <h[n,h,:], att[h,:]> ----------------
__global__ void att_coeff(const float* __restrict__ hmat,
                          const float* __restrict__ a_s,
                          const float* __restrict__ a_d,
                          float* __restrict__ asv, float* __restrict__ adv,
                          int nN, int Hh, int Cc) {
    const int tid = blockIdx.x * blockDim.x + threadIdx.x;
    if (tid >= nN * Hh) return;
    const int h = tid % Hh;
    const int n = tid / Hh;
    const float* hp = &hmat[((size_t)n * Hh + h) * Cc];
    float s = 0.f, d = 0.f;
    for (int c = 0; c < Cc; ++c) {
        const float v = hp[c];
        s += v * a_s[h * Cc + c];
        d += v * a_d[h * Cc + c];
    }
    asv[tid] = s;
    adv[tid] = d;
}

// ---------------- edge pass 1: alpha = leaky_relu(as[src]+ad[dst]); segment max ----------------
__global__ void edge_alpha_max(const long long* __restrict__ ei,
                               const float* __restrict__ asv,
                               const float* __restrict__ adv,
                               float* __restrict__ alpha,
                               float* __restrict__ amax,
                               int nE, int nN, int Hh) {
    const long tid = (long)blockIdx.x * blockDim.x + threadIdx.x;
    const long total = (long)(nE + nN) * Hh;
    if (tid >= total) return;
    const int h = (int)(tid % Hh);
    const int e = (int)(tid / Hh);
    int s, d;
    if (e < nE) { s = (int)ei[e]; d = (int)ei[nE + e]; }
    else        { s = e - nE; d = s; }                    // self-loop
    float a = asv[(size_t)s * Hh + h] + adv[(size_t)d * Hh + h];
    a = (a > 0.f) ? a : NEG_SLOPE * a;
    alpha[(size_t)e * Hh + h] = a;
    atomicMaxF(&amax[(size_t)d * Hh + h], a);
}

// ---------------- edge pass 2: ex = exp(alpha - amax[dst]); segment sum ----------------
__global__ void edge_exp_den(const long long* __restrict__ ei,
                             float* __restrict__ alpha,
                             const float* __restrict__ amax,
                             float* __restrict__ den,
                             int nE, int nN, int Hh) {
    const long tid = (long)blockIdx.x * blockDim.x + threadIdx.x;
    const long total = (long)(nE + nN) * Hh;
    if (tid >= total) return;
    const int h = (int)(tid % Hh);
    const int e = (int)(tid / Hh);
    const int d = (e < nE) ? (int)ei[nE + e] : (e - nE);
    const float ex = expf(alpha[(size_t)e * Hh + h] - amax[(size_t)d * Hh + h]);
    alpha[(size_t)e * Hh + h] = ex;                      // stash normalized numerator
    atomicAdd(&den[(size_t)d * Hh + h], ex);
}

// ---------------- edge pass 3: acc[dst] += h[src] * ex/den  (float4 per thread) ----------------
__global__ void edge_aggregate(const long long* __restrict__ ei,
                               const float* __restrict__ hmat,
                               const float* __restrict__ ex,
                               const float* __restrict__ den,
                               float* __restrict__ acc,
                               int nE, int nN, int Hh, int Cc) {
    const long tid = (long)blockIdx.x * blockDim.x + threadIdx.x;
    const int groups = Cc >> 2;
    const long total = (long)(nE + nN) * Hh * groups;
    if (tid >= total) return;
    const int g = (int)(tid % groups);
    const long t = tid / groups;
    const int h = (int)(t % Hh);
    const int e = (int)(t / Hh);
    int s, d;
    if (e < nE) { s = (int)ei[e]; d = (int)ei[nE + e]; }
    else        { s = e - nE; d = s; }
    const float w = ex[(size_t)e * Hh + h] / (den[(size_t)d * Hh + h] + 1e-16f);
    const float4 hv = *reinterpret_cast<const float4*>(&hmat[((size_t)s * Hh + h) * Cc + g * 4]);
    float* o = &acc[((size_t)d * Hh + h) * Cc + g * 4];
    atomicAdd(o + 0, hv.x * w);
    atomicAdd(o + 1, hv.y * w);
    atomicAdd(o + 2, hv.z * w);
    atomicAdd(o + 3, hv.w * w);
}

// ---------------- per-column stats: sum, sum of squares ----------------
__global__ void col_stats(const float* __restrict__ x,
                          float* __restrict__ sum, float* __restrict__ sumsq,
                          int nN, int Cc, int rowsPer) {
    const int c  = threadIdx.x;                          // blockDim.x == Cc
    const int r0 = blockIdx.x * rowsPer;
    int r1 = r0 + rowsPer; if (r1 > nN) r1 = nN;
    float s = 0.f, q = 0.f;
    for (int r = r0; r < r1; ++r) {
        const float v = x[(size_t)r * Cc + c];
        s += v; q += v * v;
    }
    atomicAdd(&sum[c], s);
    atomicAdd(&sumsq[c], q);
}

// ---------------- graph_norm finalize (+ optional ELU), in place ----------------
// var(x - s*m) = E[x^2] - 2*s*m^2 + s^2*m^2   (single-pass stats)
__global__ void norm_elu(float* __restrict__ x,
                         const float* __restrict__ sum, const float* __restrict__ sumsq,
                         const float* __restrict__ gw, const float* __restrict__ gb,
                         const float* __restrict__ gm,
                         int nN, int Cc, int doElu) {
    const long tid = (long)blockIdx.x * blockDim.x + threadIdx.x;
    if (tid >= (long)nN * Cc) return;
    const int c = (int)(tid % Cc);
    const float invN = 1.0f / (float)nN;
    const float mean = sum[c] * invN;
    const float msq  = sumsq[c] * invN;
    const float s    = gm[c];
    const float var  = msq - 2.f * s * mean * mean + s * s * mean * mean;
    const float xc   = x[tid] - s * mean;
    float y = gw[c] * xc * rsqrtf(var + EPS_) + gb[c];
    if (doElu) y = (y > 0.f) ? y : (expf(y) - 1.f);
    x[tid] = y;
}

// ---------------- small init / bias kernels ----------------
__global__ void fill_f32(float* p, float v, long n) {
    const long tid = (long)blockIdx.x * blockDim.x + threadIdx.x;
    if (tid < n) p[tid] = v;
}
__global__ void bias_rows(float* p, const float* __restrict__ b, long nN, int Cc) {
    const long tid = (long)blockIdx.x * blockDim.x + threadIdx.x;
    if (tid < nN * Cc) p[tid] = b[tid % Cc];
}
__global__ void add_bias2(float* p, const float* __restrict__ ba, const float* __restrict__ bb,
                          long nN, int Cc) {
    const long tid = (long)blockIdx.x * blockDim.x + threadIdx.x;
    if (tid < nN * Cc) p[tid] += ba[tid % Cc] + bb[tid % Cc];
}

static inline dim3 g1(long n, int b) { return dim3((unsigned)((n + b - 1) / b)); }

extern "C" void kernel_launch(void* const* d_in, const int* in_sizes, int n_in,
                              void* d_out, int out_size, void* d_ws, size_t ws_size,
                              hipStream_t stream) {
    (void)in_sizes; (void)n_in; (void)out_size; (void)ws_size;

    const float*     x  = (const float*)d_in[0];
    const long long* ei = (const long long*)d_in[1];       // int64 edge_index [2,E]
    const float* W1  = (const float*)d_in[2];
    const float* as1 = (const float*)d_in[3];
    const float* ad1 = (const float*)d_in[4];
    const float* b1  = (const float*)d_in[5];
    const float* gw1 = (const float*)d_in[6];
    const float* gb1 = (const float*)d_in[7];
    const float* gm1 = (const float*)d_in[8];
    const float* W2  = (const float*)d_in[9];
    const float* as2 = (const float*)d_in[10];
    const float* ad2 = (const float*)d_in[11];
    const float* b2  = (const float*)d_in[12];
    const float* gw2 = (const float*)d_in[13];
    const float* gb2 = (const float*)d_in[14];
    const float* gm2 = (const float*)d_in[15];
    const float* W3  = (const float*)d_in[16];
    const float* as3 = (const float*)d_in[17];
    const float* ad3 = (const float*)d_in[18];
    const float* b3  = (const float*)d_in[19];
    const float* Wres = (const float*)d_in[20];
    const float* bres = (const float*)d_in[21];
    float* out = (float*)d_out;

    // ---- workspace bump allocator (floats) ----
    float* w = (float*)d_ws;
    size_t o = 0;
    float* h1    = w + o; o += (size_t)NN * (HEADS * HID_C);       // layer1 features [N,256]
    float* acc1  = w + o; o += (size_t)NN * (HEADS * HID_C);       // layer1 aggregate / hn1
    float* ebuf  = w + o; o += (size_t)(EE + NN) * HEADS;          // per-edge alpha/ex
    float* asv   = w + o; o += (size_t)NN * HEADS;
    float* adv   = w + o; o += (size_t)NN * HEADS;
    float* amax  = w + o; o += (size_t)NN * HEADS;
    float* den   = w + o; o += (size_t)NN * HEADS;
    float* stats = w + o; o += 512;                                 // [sum 256 | sumsq 256]
    float* h2    = w + o; o += (size_t)NN * HID_C;                  // layer2/3 features
    float* acc2  = h1;                                              // reuse (h1 dead after layer1)
    float* h3    = h2;                                              // reuse (h2 dead after layer2)

    const int rowTiles   = NN / 16;                                 // 3125 (exact)
    const dim3 gemmBlk(256);                                        // 8 waves per block
    const dim3 gemmGrid((unsigned)((rowTiles + 7) / 8));            // 391 blocks
    const long nTotE = (long)EE + NN;

    // ===== residual: out = x @ Wres^T ; then out += bres + b3 =====
    gemm_bf16_wmma<IN_F, true><<<gemmGrid, gemmBlk, IN_F * OUT_C * sizeof(__bf16), stream>>>(
        x, Wres, out, NN, OUT_C);
    add_bias2<<<g1((long)NN * OUT_C, 256), 256, 0, stream>>>(out, bres, b3, NN, OUT_C);

    // ===== layer 1: GATConv(128 -> 4x64), graph_norm, elu =====
    gemm_bf16_wmma<IN_F, false><<<gemmGrid, gemmBlk,
                                  IN_F * (HEADS * HID_C) * sizeof(__bf16), stream>>>(
        x, W1, h1, NN, HEADS * HID_C);
    att_coeff<<<g1((long)NN * HEADS, 256), 256, 0, stream>>>(h1, as1, ad1, asv, adv, NN, HEADS, HID_C);
    fill_f32<<<g1((long)NN * HEADS, 256), 256, 0, stream>>>(amax, -3.0e38f, (long)NN * HEADS);
    hipMemsetAsync(den, 0, (size_t)NN * HEADS * sizeof(float), stream);
    bias_rows<<<g1((long)NN * HEADS * HID_C, 256), 256, 0, stream>>>(acc1, b1, NN, HEADS * HID_C);
    edge_alpha_max<<<g1(nTotE * HEADS, 256), 256, 0, stream>>>(ei, asv, adv, ebuf, amax, EE, NN, HEADS);
    edge_exp_den<<<g1(nTotE * HEADS, 256), 256, 0, stream>>>(ei, ebuf, amax, den, EE, NN, HEADS);
    edge_aggregate<<<g1(nTotE * HEADS * (HID_C / 4), 256), 256, 0, stream>>>(
        ei, h1, ebuf, den, acc1, EE, NN, HEADS, HID_C);
    hipMemsetAsync(stats, 0, 512 * sizeof(float), stream);
    col_stats<<<g1(NN, 128), dim3(HEADS * HID_C), 0, stream>>>(acc1, stats, stats + 256, NN, HEADS * HID_C, 128);
    norm_elu<<<g1((long)NN * HEADS * HID_C, 256), 256, 0, stream>>>(
        acc1, stats, stats + 256, gw1, gb1, gm1, NN, HEADS * HID_C, 1);

    // ===== layer 2: GATConv(256 -> 64), graph_norm, elu =====
    gemm_bf16_wmma<HEADS * HID_C, false><<<gemmGrid, gemmBlk,
                                           (HEADS * HID_C) * HID_C * sizeof(__bf16), stream>>>(
        acc1, W2, h2, NN, HID_C);
    att_coeff<<<g1((long)NN, 256), 256, 0, stream>>>(h2, as2, ad2, asv, adv, NN, 1, HID_C);
    fill_f32<<<g1((long)NN, 256), 256, 0, stream>>>(amax, -3.0e38f, (long)NN);
    hipMemsetAsync(den, 0, (size_t)NN * sizeof(float), stream);
    bias_rows<<<g1((long)NN * HID_C, 256), 256, 0, stream>>>(acc2, b2, NN, HID_C);
    edge_alpha_max<<<g1(nTotE, 256), 256, 0, stream>>>(ei, asv, adv, ebuf, amax, EE, NN, 1);
    edge_exp_den<<<g1(nTotE, 256), 256, 0, stream>>>(ei, ebuf, amax, den, EE, NN, 1);
    edge_aggregate<<<g1(nTotE * (HID_C / 4), 256), 256, 0, stream>>>(
        ei, h2, ebuf, den, acc2, EE, NN, 1, HID_C);
    hipMemsetAsync(stats, 0, 512 * sizeof(float), stream);
    col_stats<<<g1(NN, 128), dim3(HID_C), 0, stream>>>(acc2, stats, stats + 256, NN, HID_C, 128);
    norm_elu<<<g1((long)NN * HID_C, 256), 256, 0, stream>>>(
        acc2, stats, stats + 256, gw2, gb2, gm2, NN, HID_C, 1);

    // ===== layer 3: GATConv(64 -> 64), aggregate straight into out (= res + bres + b3) =====
    gemm_bf16_wmma<HID_C, false><<<gemmGrid, gemmBlk, HID_C * OUT_C * sizeof(__bf16), stream>>>(
        acc2, W3, h3, NN, OUT_C);
    att_coeff<<<g1((long)NN, 256), 256, 0, stream>>>(h3, as3, ad3, asv, adv, NN, 1, OUT_C);
    fill_f32<<<g1((long)NN, 256), 256, 0, stream>>>(amax, -3.0e38f, (long)NN);
    hipMemsetAsync(den, 0, (size_t)NN * sizeof(float), stream);
    edge_alpha_max<<<g1(nTotE, 256), 256, 0, stream>>>(ei, asv, adv, ebuf, amax, EE, NN, 1);
    edge_exp_den<<<g1(nTotE, 256), 256, 0, stream>>>(ei, ebuf, amax, den, EE, NN, 1);
    edge_aggregate<<<g1(nTotE * (OUT_C / 4), 256), 256, 0, stream>>>(
        ei, h3, ebuf, den, out, EE, NN, 1, OUT_C);
}
